// SpatialGraph_11991548691264
// MI455X (gfx1250) — matrix operand order repaired and checked by reading, not compiled, not run on hardware
//
#include <hip/hip_runtime.h>

// ---------------------------------------------------------------------------
// SpatialGraph attention for MI455X (gfx1250, wave32, WMMA bf16 16x16x32).
// B=256, C=256, H=512, N=100 boxes, scales {10,20}, IMG=224.
// Register-tiled GEMMs: 2 M-tiles (qkv) / 2x2 tiles (scores, attnv, final)
// per wave to raise WMMA density per fragment load.
// ---------------------------------------------------------------------------

typedef __attribute__((ext_vector_type(16))) __bf16 v16bf;
typedef __attribute__((ext_vector_type(8)))  float  v8f;

#define BB   256   // batch
#define CC   256   // classes / tokens
#define HH   512   // hidden
#define NBOX 100
#define WPC  17    // bitset words per class: 4 (10x10) + 13 (20x20)

__device__ __forceinline__ unsigned short f2bf(float f) {
  unsigned u = __builtin_bit_cast(unsigned, f);
  u += 0x7fffu + ((u >> 16) & 1u);          // round-to-nearest-even
  return (unsigned short)(u >> 16);
}

__device__ __forceinline__ v16bf ldfrag(const unsigned short* p) {
  return *reinterpret_cast<const v16bf*>(p);
}

__device__ __forceinline__ v8f wmma_bf16(v16bf a, v16bf b, v8f c) {
  return __builtin_amdgcn_wmma_f32_16x16x32_bf16(false, a, false, b,
                                                 (short)0, c, false, false);
}

// ---- packed-layout scatter-store helpers ----------------------------------
// A layout, per-batch tiles (M=c in 16s, K=h in 32s)
__device__ __forceinline__ void st_apackB(unsigned short* p, int b, int c, int h, float f) {
  int rm = c & 15, mtl = c >> 4, kth = h >> 5, kk = h & 31;
  int l2 = rm + 16 * ((kk >> 3) & 1);
  int e2 = (kk & 7) + ((kk >> 4) << 3);
  p[(((size_t)(b * 16 + mtl) * 16 + kth) * 32 + l2) * 16 + e2] = f2bf(f);
}
// B layout, per-batch, k=h (16 ktiles), n=c (16 ntiles)   (for K in Q@K^T)
__device__ __forceinline__ void st_bpackK(unsigned short* p, int b, int c, int h, float f) {
  int kt2 = h >> 5, nt2 = c >> 4, ki = h & 31, ni = c & 15;
  p[(((size_t)(b * 16 + kt2) * 16 + nt2) * 32 + (ni + 16 * ((ki >> 3) & 1))) * 16 + ((ki & 7) + ((ki >> 4) << 3))] = f2bf(f);
}
// B layout, per-batch, k=c (8 ktiles), n=h (32 ntiles)    (for V in attn@V)
__device__ __forceinline__ void st_bpackV(unsigned short* p, int b, int c, int h, float f) {
  int kt2 = c >> 5, nt2 = h >> 4, ki = c & 31, ni = h & 15;
  p[(((size_t)(b * 8 + kt2) * 32 + nt2) * 32 + (ni + 16 * ((ki >> 3) & 1))) * 16 + ((ki & 7) + ((ki >> 4) << 3))] = f2bf(f);
}
// A layout, global-M tiles (m in 16s over 65536 rows, K=h in 32s)
__device__ __forceinline__ void st_apackG(unsigned short* p, int m, int h, float f) {
  int mtg = m >> 4, rm = m & 15, kt2 = h >> 5, kk = h & 31;
  p[(((size_t)mtg * 16 + kt2) * 32 + (rm + 16 * ((kk >> 3) & 1))) * 16 +
    ((kk & 7) + ((kk >> 4) << 3))] = f2bf(f);
}

// ---------------------------------------------------------------------------
// K0: pack 4 weight matrices [512x512] f32 -> bf16 B-fragment layout.
// ---------------------------------------------------------------------------
__global__ __launch_bounds__(256) void k_pack_w(
    const float* __restrict__ Wq, const float* __restrict__ Wk,
    const float* __restrict__ Wv, const float* __restrict__ Wo,
    unsigned short* __restrict__ wp) {
  int idx  = blockIdx.x * 256 + threadIdx.x;     // < 4*512*512
  int w    = idx >> 18;
  int rem  = idx & 262143;
  int tile = rem >> 9;                           // (kt*32 + nt), 512 tiles
  int lane = (rem >> 4) & 31;
  int e    = rem & 15;
  int kt   = tile >> 5, nt = tile & 31;
  int k    = kt * 32 + (e & 7) + ((e >> 3) << 4) + ((lane >> 4) << 3);
  int n    = nt * 16 + (lane & 15);
  const float* W = (w == 0) ? Wq : (w == 1) ? Wk : (w == 2) ? Wv : Wo;
  wp[idx] = f2bf(W[k * HH + n]);
}

// ---------------------------------------------------------------------------
// K1: pack x [65536 x 512] f32 -> bf16 A-fragment layout.
// ---------------------------------------------------------------------------
__global__ __launch_bounds__(256) void k_pack_x(const float* __restrict__ x,
                                                unsigned short* __restrict__ xp) {
  size_t idx = (size_t)blockIdx.x * 256 + threadIdx.x;  // < 33554432
  int tile = (int)(idx >> 9);                            // mt*16 + kt
  int lane = ((int)idx >> 4) & 31;
  int e    = (int)idx & 15;
  int mt   = tile >> 4, kt = tile & 15;
  int m    = mt * 16 + (lane & 15);
  int k    = kt * 32 + (e & 7) + ((e >> 3) << 4) + ((lane >> 4) << 3);
  xp[idx]  = f2bf(x[(size_t)m * HH + k]);
}

// ---------------------------------------------------------------------------
// K2: rasterize boxes into per-class occupancy bitsets (one block per batch).
// ---------------------------------------------------------------------------
__global__ __launch_bounds__(256) void k_raster(const float* __restrict__ bboxes,
                                                const int* __restrict__ cls,
                                                unsigned* __restrict__ bits) {
  __shared__ unsigned sb[CC * WPC];
  int b = blockIdx.x;
  for (int i = threadIdx.x; i < CC * WPC; i += 256) sb[i] = 0u;
  __syncthreads();
  for (int n = threadIdx.x; n < NBOX; n += 256) {
    const float* box = bboxes + ((size_t)b * NBOX + n) * 4;
    float bx = box[0], by = box[1], bw = box[2], bh = box[3];
    int c = cls[b * NBOX + n];
    if (c < 0 || c >= CC) continue;
    float x1 = fminf(fmaxf(bx, 0.f), 223.f);
    float y1 = fminf(fmaxf(by, 0.f), 223.f);
    float x2 = fminf(fmaxf(bx + bw, 0.f), 223.f);
    float y2 = fminf(fmaxf(by + bh, 0.f), 223.f);
    {
      int gx1 = (int)(x1 * (10.f / 224.f)), gy1 = (int)(y1 * (10.f / 224.f));
      int gx2 = (int)(x2 * (10.f / 224.f)), gy2 = (int)(y2 * (10.f / 224.f));
      for (int gy = gy1; gy <= gy2; ++gy)
        for (int gx = gx1; gx <= gx2; ++gx) {
          int p = gy * 10 + gx;
          atomicOr(&sb[c * WPC + (p >> 5)], 1u << (p & 31));
        }
    }
    {
      int gx1 = (int)(x1 * (20.f / 224.f)), gy1 = (int)(y1 * (20.f / 224.f));
      int gx2 = (int)(x2 * (20.f / 224.f)), gy2 = (int)(y2 * (20.f / 224.f));
      for (int gy = gy1; gy <= gy2; ++gy)
        for (int gx = gx1; gx <= gx2; ++gx) {
          int p = gy * 20 + gx;
          atomicOr(&sb[c * WPC + 4 + (p >> 5)], 1u << (p & 31));
        }
    }
  }
  __syncthreads();
  for (int i = threadIdx.x; i < CC * WPC; i += 256)
    bits[(size_t)b * (CC * WPC) + i] = sb[i];
}

// ---------------------------------------------------------------------------
// K3: affinity via popcounts + row L1 normalization (one block per batch).
// ---------------------------------------------------------------------------
__global__ __launch_bounds__(256) void k_affinity(const unsigned* __restrict__ bits,
                                                  float* __restrict__ aff) {
  __shared__ unsigned sb[CC * WPC];
  __shared__ int s10[CC], s20[CC];
  int b = blockIdx.x, i = threadIdx.x;
  for (int t = threadIdx.x; t < CC * WPC; t += 256)
    sb[t] = bits[(size_t)b * (CC * WPC) + t];
  __syncthreads();
  unsigned di[WPC];
  int si10 = 0, si20 = 0;
  for (int w = 0; w < WPC; ++w) di[w] = sb[i * WPC + w];
  for (int w = 0; w < 4; ++w)  si10 += __popc(di[w]);
  for (int w = 4; w < WPC; ++w) si20 += __popc(di[w]);
  s10[i] = si10; s20[i] = si20;
  __syncthreads();
  float* row = aff + ((size_t)b << 16) + (size_t)i * CC;
  float rs = 0.f;
  for (int j = 0; j < CC; ++j) {
    const unsigned* dj = &sb[j * WPC];
    int g10 = 0, g20 = 0;
    for (int w = 0; w < 4; ++w)  g10 += __popc(di[w] & dj[w]);
    for (int w = 4; w < WPC; ++w) g20 += __popc(di[w] & dj[w]);
    float l1a = (float)(si10 + s10[j] - 2 * g10);
    float l1b = (float)(si20 + s20[j] - 2 * g20);
    float a = 0.5f * (1.f - l1a * (1.f / 100.f)) + 0.5f * (1.f - l1b * (1.f / 400.f));
    rs += fabsf(a);
    row[j] = a;
  }
  float inv = 1.f / fmaxf(rs, 1e-12f);
  for (int j = 0; j < CC; ++j) row[j] *= inv;
}

// ---------------------------------------------------------------------------
// K4: fused Q/K/V projection, 2 M-tiles per wave (6 WMMAs per K-step, B frags
// shared across M). Outputs pre-packed for the downstream GEMMs.
// ---------------------------------------------------------------------------
__global__ __launch_bounds__(256) void k_qkv(
    const unsigned short* __restrict__ xp, const unsigned short* __restrict__ wp,
    const float* __restrict__ bq, const float* __restrict__ bk,
    const float* __restrict__ bv,
    unsigned short* __restrict__ qp, unsigned short* __restrict__ kp,
    unsigned short* __restrict__ vp) {
  int wi   = blockIdx.x * 8 + (threadIdx.x >> 5);  // < 2048*32
  int lane = threadIdx.x & 31;
  int mt2  = wi >> 5;                              // pair of M tiles, 0..2047
  int nt   = wi & 31;                              // N tile, 0..31
  const unsigned short* wq = wp;
  const unsigned short* wk = wp + 262144;
  const unsigned short* wv = wp + 524288;
  v8f z = {0,0,0,0,0,0,0,0};
  v8f aq[2] = {z, z}, ak[2] = {z, z}, av[2] = {z, z};
#pragma unroll
  for (int kt = 0; kt < 16; ++kt) {
    const unsigned short* ap0 = xp + (((size_t)(mt2 * 2) * 16 + kt) * 32 + lane) * 16;
    const unsigned short* ap1 = xp + (((size_t)(mt2 * 2 + 1) * 16 + kt) * 32 + lane) * 16;
    __builtin_prefetch(ap0 + 512, 0, 0);
    v16bf A0 = ldfrag(ap0);
    v16bf A1 = ldfrag(ap1);
    v16bf Bq = ldfrag(wq + (((size_t)kt * 32 + nt) * 32 + lane) * 16);
    v16bf Bk = ldfrag(wk + (((size_t)kt * 32 + nt) * 32 + lane) * 16);
    v16bf Bv = ldfrag(wv + (((size_t)kt * 32 + nt) * 32 + lane) * 16);
    aq[0] = wmma_bf16(A0, Bq, aq[0]);  aq[1] = wmma_bf16(A1, Bq, aq[1]);
    ak[0] = wmma_bf16(A0, Bk, ak[0]);  ak[1] = wmma_bf16(A1, Bk, ak[1]);
    av[0] = wmma_bf16(A0, Bv, av[0]);  av[1] = wmma_bf16(A1, Bv, av[1]);
  }
  int hl = lane >> 4, ln = lane & 15;
  int h = nt * 16 + ln;
  float biasq = bq[h], biask = bk[h], biasv = bv[h];
#pragma unroll
  for (int im = 0; im < 2; ++im) {
#pragma unroll
    for (int r = 0; r < 8; ++r) {
      int m = (mt2 * 2 + im) * 16 + r + 8 * hl;   // global row
      int b = m >> 8, c = m & 255;
      st_apackB(qp, b, c, h, aq[im][r] + biasq);
      st_bpackK(kp, b, c, h, ak[im][r] + biask);
      st_bpackV(vp, b, c, h, av[im][r] + biasv);
    }
  }
}

// ---------------------------------------------------------------------------
// K5: scores = (Q @ K^T) / sqrt(H) * aff, 2x2 tiles per wave, f32 output.
// ---------------------------------------------------------------------------
__global__ __launch_bounds__(256) void k_scores(
    const unsigned short* __restrict__ qp, const unsigned short* __restrict__ kp,
    const float* __restrict__ aff, float* __restrict__ scores) {
  int wi = blockIdx.x * 8 + (threadIdx.x >> 5);  // < 256*64
  int lane = threadIdx.x & 31;
  int b = wi >> 6;
  int t = wi & 63;
  int mt2 = t >> 3, nt2 = t & 7;                 // 2x2 super-tile coords
  v8f z = {0,0,0,0,0,0,0,0};
  v8f acc[2][2] = {{z, z}, {z, z}};
#pragma unroll
  for (int kt = 0; kt < 16; ++kt) {
    v16bf A0 = ldfrag(qp + (((size_t)(b * 16 + mt2 * 2)     * 16 + kt) * 32 + lane) * 16);
    v16bf A1 = ldfrag(qp + (((size_t)(b * 16 + mt2 * 2 + 1) * 16 + kt) * 32 + lane) * 16);
    v16bf B0 = ldfrag(kp + (((size_t)(b * 16 + kt) * 16 + nt2 * 2)     * 32 + lane) * 16);
    v16bf B1 = ldfrag(kp + (((size_t)(b * 16 + kt) * 16 + nt2 * 2 + 1) * 32 + lane) * 16);
    acc[0][0] = wmma_bf16(A0, B0, acc[0][0]);  acc[0][1] = wmma_bf16(A0, B1, acc[0][1]);
    acc[1][0] = wmma_bf16(A1, B0, acc[1][0]);  acc[1][1] = wmma_bf16(A1, B1, acc[1][1]);
  }
  int hl = lane >> 4, ln = lane & 15;
  const float sc = 0.04419417382415922f;  // 1/sqrt(512)
#pragma unroll
  for (int im = 0; im < 2; ++im)
#pragma unroll
    for (int in = 0; in < 2; ++in)
#pragma unroll
      for (int r = 0; r < 8; ++r) {
        int c = (mt2 * 2 + im) * 16 + r + 8 * hl;
        int d = (nt2 * 2 + in) * 16 + ln;
        size_t o = ((size_t)b << 16) + (size_t)c * 256 + d;
        scores[o] = acc[im][in][r] * sc * aff[o];
      }
}

// ---------------------------------------------------------------------------
// K6: softmax over each 256-wide row (one wave per row), write bf16 in
// per-batch A-fragment layout for the attn@V GEMM.
// ---------------------------------------------------------------------------
__global__ __launch_bounds__(256) void k_softmax(const float* __restrict__ scores,
                                                 unsigned short* __restrict__ attnp) {
  int row  = blockIdx.x * 8 + (threadIdx.x >> 5);  // < 65536 (b*256+c)
  int lane = threadIdx.x & 31;
  int b = row >> 8, c = row & 255;
  const float* base = scores + (size_t)row * 256;
  float4 v0 = reinterpret_cast<const float4*>(base)[lane * 2];
  float4 v1 = reinterpret_cast<const float4*>(base)[lane * 2 + 1];
  float v[8] = {v0.x, v0.y, v0.z, v0.w, v1.x, v1.y, v1.z, v1.w};
  float mx = v[0];
  for (int t = 1; t < 8; ++t) mx = fmaxf(mx, v[t]);
  for (int off = 16; off > 0; off >>= 1) mx = fmaxf(mx, __shfl_xor(mx, off, 32));
  float sum = 0.f;
  for (int t = 0; t < 8; ++t) { v[t] = __expf(v[t] - mx); sum += v[t]; }
  for (int off = 16; off > 0; off >>= 1) sum += __shfl_xor(sum, off, 32);
  float inv = 1.f / sum;
  int rm = c & 15, mtl = c >> 4;
#pragma unroll
  for (int t = 0; t < 8; ++t) {
    int d = lane * 8 + t;
    int kt = d >> 5, kk = d & 31;
    int l2 = rm + 16 * ((kk >> 3) & 1);
    int e2 = (kk & 7) + ((kk >> 4) << 3);
    attnp[(((size_t)(b * 16 + mtl) * 8 + kt) * 32 + l2) * 16 + e2] = f2bf(v[t] * inv);
  }
}

// ---------------------------------------------------------------------------
// K7: out_attn = attn @ V, 2x2 tiles per wave, written A-packed for K8.
// ---------------------------------------------------------------------------
__global__ __launch_bounds__(256) void k_attnv(
    const unsigned short* __restrict__ attnp, const unsigned short* __restrict__ vp,
    unsigned short* __restrict__ oap) {
  int wi = blockIdx.x * 8 + (threadIdx.x >> 5);  // < 256*128
  int lane = threadIdx.x & 31;
  int b = wi >> 7;
  int t = wi & 127;
  int mt2 = t >> 4, nt2 = t & 15;
  v8f z = {0,0,0,0,0,0,0,0};
  v8f acc[2][2] = {{z, z}, {z, z}};
#pragma unroll
  for (int kt = 0; kt < 8; ++kt) {
    v16bf A0 = ldfrag(attnp + (((size_t)(b * 16 + mt2 * 2)     * 8 + kt) * 32 + lane) * 16);
    v16bf A1 = ldfrag(attnp + (((size_t)(b * 16 + mt2 * 2 + 1) * 8 + kt) * 32 + lane) * 16);
    v16bf B0 = ldfrag(vp + (((size_t)(b * 8 + kt) * 32 + nt2 * 2)     * 32 + lane) * 16);
    v16bf B1 = ldfrag(vp + (((size_t)(b * 8 + kt) * 32 + nt2 * 2 + 1) * 32 + lane) * 16);
    acc[0][0] = wmma_bf16(A0, B0, acc[0][0]);  acc[0][1] = wmma_bf16(A0, B1, acc[0][1]);
    acc[1][0] = wmma_bf16(A1, B0, acc[1][0]);  acc[1][1] = wmma_bf16(A1, B1, acc[1][1]);
  }
  int hl = lane >> 4, ln = lane & 15;
#pragma unroll
  for (int im = 0; im < 2; ++im)
#pragma unroll
    for (int in = 0; in < 2; ++in)
#pragma unroll
      for (int r = 0; r < 8; ++r) {
        int c = (mt2 * 2 + im) * 16 + r + 8 * hl;
        int h = (nt2 * 2 + in) * 16 + ln;
        st_apackG(oap, b * 256 + c, h, acc[im][in][r]);
      }
}

// ---------------------------------------------------------------------------
// K8: final = out_attn @ Wo + bo, 2x2 tiles per wave, f32 output [B*C, H].
// ---------------------------------------------------------------------------
__global__ __launch_bounds__(256) void k_final(
    const unsigned short* __restrict__ oap, const unsigned short* __restrict__ wop,
    const float* __restrict__ bo, float* __restrict__ out) {
  int wi = blockIdx.x * 8 + (threadIdx.x >> 5);  // < 2048*16
  int lane = threadIdx.x & 31;
  int mt2 = wi >> 4, nt2 = wi & 15;
  v8f z = {0,0,0,0,0,0,0,0};
  v8f acc[2][2] = {{z, z}, {z, z}};
#pragma unroll
  for (int kt = 0; kt < 16; ++kt) {
    const unsigned short* ap0 = oap + (((size_t)(mt2 * 2) * 16 + kt) * 32 + lane) * 16;
    const unsigned short* ap1 = oap + (((size_t)(mt2 * 2 + 1) * 16 + kt) * 32 + lane) * 16;
    __builtin_prefetch(ap0 + 512, 0, 0);
    v16bf A0 = ldfrag(ap0);
    v16bf A1 = ldfrag(ap1);
    v16bf B0 = ldfrag(wop + (((size_t)kt * 32 + nt2 * 2)     * 32 + lane) * 16);
    v16bf B1 = ldfrag(wop + (((size_t)kt * 32 + nt2 * 2 + 1) * 32 + lane) * 16);
    acc[0][0] = wmma_bf16(A0, B0, acc[0][0]);  acc[0][1] = wmma_bf16(A0, B1, acc[0][1]);
    acc[1][0] = wmma_bf16(A1, B0, acc[1][0]);  acc[1][1] = wmma_bf16(A1, B1, acc[1][1]);
  }
  int hl = lane >> 4, ln = lane & 15;
#pragma unroll
  for (int im = 0; im < 2; ++im)
#pragma unroll
    for (int in = 0; in < 2; ++in) {
      int h = (nt2 * 2 + in) * 16 + ln;
      float bias = bo[h];
#pragma unroll
      for (int r = 0; r < 8; ++r) {
        int m = (mt2 * 2 + im) * 16 + r + 8 * hl;
        out[(size_t)m * HH + h] = acc[im][in][r] + bias;
      }
    }
}

// ---------------------------------------------------------------------------
// Launch. Workspace layout (byte offsets), lifetime-based reuse:
//   [0,64M)    x_pack bf16   -> reused as scores f32
//   [64,128M)  Q pack bf16   -> reused as attn pack bf16
//   [128,192M) K pack bf16   -> reused as out_attn pack bf16
//   [192,256M) V pack bf16
//   [256M,+2M) packed weights, then aff f32 (64MB), then bitsets (~4.5MB)
// ---------------------------------------------------------------------------
extern "C" void kernel_launch(void* const* d_in, const int* in_sizes, int n_in,
                              void* d_out, int out_size, void* d_ws, size_t ws_size,
                              hipStream_t stream) {
  const float* x      = (const float*)d_in[0];
  const float* bboxes = (const float*)d_in[1];
  const int*   cls    = (const int*)d_in[2];
  const float* Wq = (const float*)d_in[3];
  const float* bq = (const float*)d_in[4];
  const float* Wk = (const float*)d_in[5];
  const float* bk = (const float*)d_in[6];
  const float* Wv = (const float*)d_in[7];
  const float* bv = (const float*)d_in[8];
  const float* Wo = (const float*)d_in[9];
  const float* bo = (const float*)d_in[10];
  float* out = (float*)d_out;

  char* ws = (char*)d_ws;
  const size_t MB64 = 67108864ull;
  unsigned short* xp     = (unsigned short*)(ws);                 // 64MB
  float*          scores = (float*)(ws);                          // reuse
  unsigned short* qp     = (unsigned short*)(ws + MB64);          // 64MB
  unsigned short* attnp  = (unsigned short*)(ws + MB64);          // reuse
  unsigned short* kp     = (unsigned short*)(ws + 2 * MB64);      // 64MB
  unsigned short* oap    = (unsigned short*)(ws + 2 * MB64);      // reuse
  unsigned short* vp     = (unsigned short*)(ws + 3 * MB64);      // 64MB
  unsigned short* wp     = (unsigned short*)(ws + 4 * MB64);      // 2MB
  float*          aff    = (float*)(ws + 4 * MB64 + 2097152ull);  // 64MB
  unsigned*       bits   = (unsigned*)(ws + 5 * MB64 + 2097152ull); // ~4.5MB
  unsigned short* wop    = wp + 3 * 262144;

  k_pack_w  <<<4096,   256, 0, stream>>>(Wq, Wk, Wv, Wo, wp);
  k_pack_x  <<<131072, 256, 0, stream>>>(x, xp);
  k_raster  <<<256,    256, 0, stream>>>(bboxes, cls, bits);
  k_affinity<<<256,    256, 0, stream>>>(bits, aff);
  k_qkv     <<<8192,   256, 0, stream>>>(xp, wp, bq, bk, bv, qp, kp, vp);
  k_scores  <<<2048,   256, 0, stream>>>(qp, kp, aff, scores);
  k_softmax <<<8192,   256, 0, stream>>>(scores, attnp);
  k_attnv   <<<4096,   256, 0, stream>>>(attnp, vp, oap);
  k_final   <<<4096,   256, 0, stream>>>(oap, wop, bo, out);

  (void)in_sizes; (void)n_in; (void)out_size; (void)ws_size;
}
